// Mamba2_69604239999535
// MI455X (gfx1250) — compile-verified, matrix operand order checked by reading, not run on hardware
//
#include <hip/hip_runtime.h>
#include <hip/hip_bf16.h>
#include <math.h>

// ---------------- problem constants ----------------
#define HIDDEN   2048
#define INTER    4096
#define HEADS    32
#define HEAD_DIM 128
#define STATE    128
#define CONV_KW  4
#define CHUNK    256
#define CONV_DIM (INTER + 2*STATE)          // 4352
#define NPROJ    (INTER + CONV_DIM + HEADS) // 8480
#define NPROJ_PAD 8576                      // 67 * 128
#define BATCH    2
#define SEQL     4096
#define NCHUNK   (SEQL / CHUNK)             // 16
#define MTOT     (BATCH * SEQL)             // 8192
#define EPSV     1e-5f

typedef __bf16 bf16;
typedef __attribute__((ext_vector_type(16))) __bf16 v16bf;
typedef __attribute__((ext_vector_type(8)))  __bf16 v8bf;
typedef __attribute__((ext_vector_type(8)))  float  v8f;

// ---------------- WMMA helpers ----------------
static __device__ __forceinline__ v8f wmma_bf16(v16bf a, v16bf b, v8f c) {
  // D = A(16x32) * B(32x16) + C, fp32 accumulate
  return __builtin_amdgcn_wmma_f32_16x16x32_bf16(false, a, false, b, (short)0, c,
                                                 false, false);
}

// Load a 16x32 A-operand fragment (or B operand from N-major storage) from a
// row-major [row][ld] bf16 array.  ISA layout: lane m (0..15) holds row m,
// K = 0..7 then 16..23 ; lane m+16 holds row m, K = 8..15 then 24..31.
// Both 8-element runs are contiguous 16B -> two b128 loads.
static __device__ __forceinline__ v16bf frag_load(const bf16* p, int ld,
                                                  int row_base, int k_base) {
  int lane = threadIdx.x & 31;
  int r  = row_base + (lane & 15);
  int k0 = k_base + ((lane >> 4) << 3);
  const bf16* q = p + (size_t)r * (size_t)ld + k0;
  v8bf lo = *(const v8bf*)(q);
  v8bf hi = *(const v8bf*)(q + 16);
  v16bf f;
#pragma unroll
  for (int i = 0; i < 8; ++i) { f[i] = lo[i]; f[i + 8] = hi[i]; }
  return f;
}

static __device__ __forceinline__ v8f v8f_zero() {
  v8f z = {0.f,0.f,0.f,0.f,0.f,0.f,0.f,0.f};
  return z;
}
static __device__ __forceinline__ float silu(float x) {
  return x / (1.f + __expf(-x));
}

// ---- CDNA5 async LDS DMA (ASYNCcnt-tracked, bypasses VGPRs) ----
// LDS operand: low 32 bits of the flat address are the LDS byte offset.
static __device__ __forceinline__ void async_b128(const bf16* g, const bf16* l) {
  unsigned lds = (unsigned)(uintptr_t)l;
  asm volatile("global_load_async_to_lds_b128 %0, %1, off"
               :: "v"(lds), "v"(g) : "memory");
}
static __device__ __forceinline__ void wait_async0() {
  asm volatile("s_wait_asynccnt 0" ::: "memory");
}

// ---------------- elementwise packing ----------------
__global__ __launch_bounds__(256) void k_f32_to_bf16(const float* __restrict__ s,
                                                     bf16* __restrict__ d, long n) {
  long i = (long)blockIdx.x * 256 + threadIdx.x;
  if (i < n) d[i] = (bf16)s[i];
}

__global__ __launch_bounds__(256) void k_pack_w_in(const float* __restrict__ w,
                                                   bf16* __restrict__ d) {
  long i = (long)blockIdx.x * 256 + threadIdx.x;
  long n = (long)NPROJ_PAD * HIDDEN;
  if (i >= n) return;
  long row = i / HIDDEN, col = i - row * HIDDEN;
  d[i] = (row < NPROJ) ? (bf16)w[row * HIDDEN + col] : (bf16)0.0f;
}

// ---------------- tiled BF16 WMMA GEMM:  C = A[MxK] * W[NxK]^T ----------------
// 256x128 block tile, 8 waves of 64x64, K panels of 32, double-buffered LDS
// staged with async LDS DMA.  16 WMMA : 16 ds_load_b128 per wave per K-step.
#define GT_M 256
#define GT_N 128
#define GT_K 32
#define GLD  40   // LDS row stride (bf16), mult of 8 -> 16B aligned b128 reads

__global__ __launch_bounds__(256)
void k_gemm(const bf16* __restrict__ A, const bf16* __restrict__ W,
            int M, int N, int K, int mode,
            float* __restrict__ outF,      // mode 0
            bf16* __restrict__ outGate,    // mode 1: cols [0,4096)
            bf16* __restrict__ outHBC,     // mode 1: cols [4096,8448)
            float* __restrict__ outDt)     // mode 1: cols [8448,8480)
{
  __shared__ bf16 As[2][GT_M * GLD];   // 2 x 20 KB
  __shared__ bf16 Ws[2][GT_N * GLD];   // 2 x 10 KB
  const int tid  = threadIdx.x;
  const int wid  = tid >> 5;
  const int lane = tid & 31;
  const int m0 = blockIdx.y * GT_M;
  const int n0 = blockIdx.x * GT_N;
  const int wm = (wid >> 1) * 64;   // 0,64,128,192
  const int wn = (wid & 1) * 64;    // 0,64

  v8f acc[4][4];
#pragma unroll
  for (int i = 0; i < 4; ++i)
#pragma unroll
    for (int j = 0; j < 4; ++j) acc[i][j] = v8f_zero();

  // async-stage one 256x32 A panel + 128x32 W panel into LDS buffer `buf`
  auto stage = [&](int buf, int k0) {
#pragma unroll
    for (int i = 0; i < 4; ++i) {              // 1024 16B chunks of A
      int cid = tid + i * 256;
      int row = cid >> 2;
      int kc  = (cid & 3) * 8;
      async_b128(&A[(size_t)(m0 + row) * K + k0 + kc], &As[buf][row * GLD + kc]);
    }
#pragma unroll
    for (int i = 0; i < 2; ++i) {              // 512 16B chunks of W
      int cid = tid + i * 256;
      int row = cid >> 2;
      int kc  = (cid & 3) * 8;
      async_b128(&W[(size_t)(n0 + row) * K + k0 + kc], &Ws[buf][row * GLD + kc]);
    }
  };

  const int nk = K / GT_K;
  stage(0, 0);
  wait_async0();
  __syncthreads();

  for (int kk = 0; kk < nk; ++kk) {
    const int cur = kk & 1;
    if (kk + 1 < nk) stage(cur ^ 1, (kk + 1) * GT_K);  // overlap with compute
    if (kk + 2 < nk) {                                  // warm L2/WGP$ 2 ahead
      __builtin_prefetch(&A[(size_t)(m0 + tid) * K + (kk + 2) * GT_K], 0, 3);
      __builtin_prefetch(&W[(size_t)(n0 + (tid & 127)) * K + (kk + 2) * GT_K], 0, 3);
    }
    v16bf af[4];
#pragma unroll
    for (int tm = 0; tm < 4; ++tm)
      af[tm] = frag_load(&As[cur][0], GLD, wm + 16 * tm, 0);
#pragma unroll
    for (int tn = 0; tn < 4; ++tn) {
      v16bf bfr = frag_load(&Ws[cur][0], GLD, wn + 16 * tn, 0);
#pragma unroll
      for (int tm = 0; tm < 4; ++tm)
        acc[tm][tn] = wmma_bf16(af[tm], bfr, acc[tm][tn]);
    }
    wait_async0();       // our async writes for buf^1 have landed
    __syncthreads();     // everyone done reading cur & staging next
  }

  const int hi2 = lane >> 4, ln = lane & 15;
#pragma unroll
  for (int tm = 0; tm < 4; ++tm)
#pragma unroll
    for (int tn = 0; tn < 4; ++tn)
#pragma unroll
      for (int v = 0; v < 8; ++v) {
        int m = m0 + wm + 16 * tm + 8 * hi2 + v;
        int n = n0 + wn + 16 * tn + ln;
        float val = acc[tm][tn][v];
        if (mode == 0) {
          if (m < M && n < N) outF[(size_t)m * N + n] = val;
        } else {
          if (n < INTER)                outGate[(size_t)m * INTER + n] = (bf16)val;
          else if (n < INTER + CONV_DIM) outHBC[(size_t)m * CONV_DIM + (n - INTER)] = (bf16)val;
          else if (n < NPROJ)            outDt[(size_t)m * HEADS + (n - INTER - CONV_DIM)] = val;
        }
      }
}

// ---------------- dt: softplus/clip, Ad cumsum per chunk ----------------
__global__ __launch_bounds__(256)
void k_dt(const float* __restrict__ dtraw, const float* __restrict__ A_log,
          const float* __restrict__ dt_bias,
          float* __restrict__ dtv, float* __restrict__ Acum, float* __restrict__ csum)
{
  int bid = blockIdx.x;                  // (b, h, c)
  int c = bid % NCHUNK;
  int h = (bid / NCHUNK) % HEADS;
  int b = bid / (NCHUNK * HEADS);
  int l = threadIdx.x;
  size_t m = (size_t)b * SEQL + c * CHUNK + l;
  float x = dtraw[m * HEADS + h] + dt_bias[h];
  float dt = (x > 20.f) ? x : log1pf(__expf(x));
  dt = fminf(fmaxf(dt, 0.0f), 100.0f);
  dtv[m * HEADS + h] = dt;
  float Ah = -__expf(A_log[h]);
  __shared__ float sc[CHUNK];
  sc[l] = Ah * dt;
  __syncthreads();
  for (int off = 1; off < CHUNK; off <<= 1) {   // inclusive scan
    float t = (l >= off) ? sc[l - off] : 0.f;
    __syncthreads();
    sc[l] += t;
    __syncthreads();
  }
  size_t base = (((size_t)b * HEADS + h) * NCHUNK + c) * CHUNK;
  Acum[base + l] = sc[l];
  if (l == CHUNK - 1) csum[((size_t)b * HEADS + h) * NCHUNK + c] = sc[l];
}

// ---------------- depthwise causal conv + SiLU + operand prep ----------------
__global__ __launch_bounds__(256)
void k_conv(const bf16* __restrict__ hbc, const float* __restrict__ convw,
            const float* __restrict__ convb, const float* __restrict__ dtv,
            bf16* __restrict__ xs, bf16* __restrict__ xdtT,
            bf16* __restrict__ Bn, bf16* __restrict__ Cn)
{
  int m = blockIdx.x;
  int b = m / SEQL, s = m % SEQL;
  int c = s / CHUNK, l = s % CHUNK;
  for (int ch = threadIdx.x; ch < CONV_DIM; ch += 256) {
    float acc = convb[ch];
#pragma unroll
    for (int k = 0; k < CONV_KW; ++k) {
      int ss = s - (CONV_KW - 1) + k;
      if (ss >= 0)
        acc += (float)hbc[((size_t)b * SEQL + ss) * CONV_DIM + ch] * convw[ch * CONV_KW + k];
    }
    float val = silu(acc);
    if (ch < INTER) {
      xs[(size_t)m * INTER + ch] = (bf16)val;
      int h = ch >> 7, p = ch & 127;
      float xdt = val * dtv[(size_t)m * HEADS + h];
      size_t idx = ((((size_t)b * HEADS + h) * NCHUNK + c) * HEAD_DIM + p) * CHUNK + l;
      xdtT[idx] = (bf16)xdt;                       // [b][h][c][p][l]
    } else if (ch < INTER + STATE) {
      Bn[(size_t)m * STATE + (ch - INTER)] = (bf16)val;
    } else {
      Cn[(size_t)m * STATE + (ch - INTER - STATE)] = (bf16)val;
    }
  }
}

// ---------------- chunk-local states: S[p][n] = sum_l xdt[l,p]*decay[l]*B[l,n] --
#define LDB (CHUNK + 8)   // 264, mult of 8
__global__ __launch_bounds__(256)
void k_states(const bf16* __restrict__ xdtT, const bf16* __restrict__ Bn,
              const float* __restrict__ Acum, float* __restrict__ Sloc)
{
  int bid = blockIdx.x;                   // (b, c, h)
  int h = bid % HEADS;
  int c = (bid / HEADS) % NCHUNK;
  int b = bid / (HEADS * NCHUNK);
  __shared__ float ac[CHUNK];
  __shared__ bf16  BdT[STATE * LDB];
  int tid = threadIdx.x;
  size_t abase = (((size_t)b * HEADS + h) * NCHUNK + c) * CHUNK;
  ac[tid] = Acum[abase + tid];
  __syncthreads();
  float atot = ac[CHUNK - 1];
  size_t bbase = ((size_t)b * SEQL + c * CHUNK) * STATE;
  for (int idx = tid; idx < CHUNK * STATE; idx += 256) {
    int l = idx >> 7, n = idx & 127;
    float v = (float)Bn[bbase + (size_t)l * STATE + n] * __expf(atot - ac[l]);
    BdT[n * LDB + l] = (bf16)v;           // transposed, decay folded in
  }
  __syncthreads();

  int wid = tid >> 5;
  int p0 = wid * 16;
  const bf16* xb = xdtT + (((size_t)b * HEADS + h) * NCHUNK + c) * (size_t)(HEAD_DIM * CHUNK);
  v8f acc[8];
#pragma unroll
  for (int i = 0; i < 8; ++i) acc[i] = v8f_zero();
  for (int kt = 0; kt < CHUNK; kt += 32) {
    v16bf a = frag_load(xb, CHUNK, p0, kt);
#pragma unroll
    for (int nt = 0; nt < 8; ++nt) {
      v16bf bb = frag_load(BdT, LDB, nt * 16, kt);
      acc[nt] = wmma_bf16(a, bb, acc[nt]);
    }
  }
  int lane = tid & 31, hi2 = lane >> 4, ln = lane & 15;
  size_t sb = ((((size_t)b * NCHUNK + c) * HEADS + h) * HEAD_DIM) * STATE;
#pragma unroll
  for (int nt = 0; nt < 8; ++nt)
#pragma unroll
    for (int v = 0; v < 8; ++v) {
      int p = p0 + 8 * hi2 + v;
      int n = nt * 16 + ln;
      Sloc[sb + (size_t)p * STATE + n] = acc[nt][v];
    }
}

// ---------------- inter-chunk recurrence ----------------
__global__ __launch_bounds__(256)
void k_chain(const float* __restrict__ Sloc, const float* __restrict__ csum,
             bf16* __restrict__ Sin)
{
  int bid  = blockIdx.x;                  // (b, h, part)
  int part = bid & 7;
  int h = (bid >> 3) % HEADS;
  int b = (bid >> 3) / HEADS;
  int e0 = part * 2048 + threadIdx.x;
  float carry[8];
#pragma unroll
  for (int i = 0; i < 8; ++i) carry[i] = 0.f;
  for (int z = 0; z < NCHUNK; ++z) {
    size_t inb = (((size_t)b * NCHUNK + z) * HEADS + h) * (size_t)(HEAD_DIM * STATE);
#pragma unroll
    for (int i = 0; i < 8; ++i) Sin[inb + e0 + i * 256] = (bf16)carry[i];
    float dk = __expf(csum[((size_t)b * HEADS + h) * NCHUNK + z]);
#pragma unroll
    for (int i = 0; i < 8; ++i)
      carry[i] = carry[i] * dk + Sloc[inb + e0 + i * 256];
  }
}

// ---------------- per-chunk output: Y_off + Y_diag + D*xs ----------------
__global__ __launch_bounds__(256)
void k_ychunk(const bf16* __restrict__ Cn, const bf16* __restrict__ Bn,
              const bf16* __restrict__ Sin, const bf16* __restrict__ xdtT,
              const bf16* __restrict__ xs, const float* __restrict__ Acum,
              const float* __restrict__ Dv, bf16* __restrict__ Y)
{
  int bid = blockIdx.x;                   // (b, c, h)
  int h = bid % HEADS;
  int c = (bid / HEADS) % NCHUNK;
  int b = bid / (HEADS * NCHUNK);
  int tid = threadIdx.x, wid = tid >> 5, lane = tid & 31;
  int hi2 = lane >> 4, ln = lane & 15;
  __shared__ float ac[CHUNK];
  __shared__ bf16  Mp[8][32 * 40];        // per-wave 32x32 M panel, ld=40

  size_t abase = (((size_t)b * HEADS + h) * NCHUNK + c) * CHUNK;
  ac[tid] = Acum[abase + tid];
  __syncthreads();

  int l0 = wid * 32;                      // wave owns 32 l-rows
  const bf16* Cb = Cn + ((size_t)b * SEQL + c * CHUNK) * STATE;
  const bf16* Bb = Bn + ((size_t)b * SEQL + c * CHUNK) * STATE;
  const bf16* Sb = Sin + (((size_t)b * NCHUNK + c) * HEADS + h) * (size_t)(HEAD_DIM * STATE);
  const bf16* Xb = xdtT + (((size_t)b * HEADS + h) * NCHUNK + c) * (size_t)(HEAD_DIM * CHUNK);

  v8f acc[2][8];
#pragma unroll
  for (int i = 0; i < 2; ++i)
#pragma unroll
    for (int j = 0; j < 8; ++j) acc[i][j] = v8f_zero();

  // ---- Y_off = C(256x128) * Sin(p,n)^T, scaled by exp(Acum[l]) ----
#pragma unroll
  for (int tm = 0; tm < 2; ++tm)
    for (int kt = 0; kt < STATE; kt += 32) {
      v16bf a = frag_load(Cb, STATE, l0 + 16 * tm, kt);
#pragma unroll
      for (int tn = 0; tn < 8; ++tn) {
        v16bf bb = frag_load(Sb, STATE, tn * 16, kt);
        acc[tm][tn] = wmma_bf16(a, bb, acc[tm][tn]);
      }
    }
#pragma unroll
  for (int tm = 0; tm < 2; ++tm)
#pragma unroll
    for (int tn = 0; tn < 8; ++tn)
#pragma unroll
      for (int v = 0; v < 8; ++v) {
        int l = l0 + 16 * tm + 8 * hi2 + v;
        acc[tm][tn][v] *= __expf(ac[l]);
      }

  // ---- Y_diag: 32-wide s panels; G = C*B^T -> mask*L -> bf16 -> WMMA vs xdtT --
  bf16* mp = &Mp[wid][0];
  for (int sb = 0; sb < CHUNK; sb += 32) {
    v8f g[2][2];
#pragma unroll
    for (int i = 0; i < 2; ++i)
#pragma unroll
      for (int j = 0; j < 2; ++j) g[i][j] = v8f_zero();
    for (int kt = 0; kt < STATE; kt += 32) {
      v16bf a0 = frag_load(Cb, STATE, l0, kt);
      v16bf a1 = frag_load(Cb, STATE, l0 + 16, kt);
      v16bf b0 = frag_load(Bb, STATE, sb, kt);
      v16bf b1 = frag_load(Bb, STATE, sb + 16, kt);
      g[0][0] = wmma_bf16(a0, b0, g[0][0]);
      g[0][1] = wmma_bf16(a0, b1, g[0][1]);
      g[1][0] = wmma_bf16(a1, b0, g[1][0]);
      g[1][1] = wmma_bf16(a1, b1, g[1][1]);
    }
#pragma unroll
    for (int tm = 0; tm < 2; ++tm)
#pragma unroll
      for (int tn = 0; tn < 2; ++tn)
#pragma unroll
        for (int v = 0; v < 8; ++v) {
          int lr = 16 * tm + 8 * hi2 + v;     // local row
          int sc2 = 16 * tn + ln;             // local col
          int l = l0 + lr, s = sb + sc2;
          float val = (s <= l) ? g[tm][tn][v] * __expf(ac[l] - ac[s]) : 0.f;
          mp[lr * 40 + sc2] = (bf16)val;
        }
    __syncthreads();
    v16bf ma0 = frag_load(mp, 40, 0, 0);
    v16bf ma1 = frag_load(mp, 40, 16, 0);
#pragma unroll
    for (int tn = 0; tn < 8; ++tn) {
      v16bf bb = frag_load(Xb, CHUNK, tn * 16, sb);
      acc[0][tn] = wmma_bf16(ma0, bb, acc[0][tn]);
      acc[1][tn] = wmma_bf16(ma1, bb, acc[1][tn]);
    }
    __syncthreads();
  }

  // ---- epilogue: + D[h]*xs, store bf16 ----
  float dh = Dv[h];
  size_t ybase = (size_t)b * SEQL + c * CHUNK;
#pragma unroll
  for (int tm = 0; tm < 2; ++tm)
#pragma unroll
    for (int tn = 0; tn < 8; ++tn)
#pragma unroll
      for (int v = 0; v < 8; ++v) {
        int l = l0 + 16 * tm + 8 * hi2 + v;
        int p = tn * 16 + ln;
        size_t row = ybase + l;
        float xsv = (float)xs[row * INTER + h * HEAD_DIM + p];
        float val = acc[tm][tn][v] + dh * xsv;
        Y[row * INTER + h * HEAD_DIM + p] = (bf16)val;
      }
}

// ---------------- gate * SiLU + RMS norm ----------------
__global__ __launch_bounds__(256)
void k_norm(const bf16* __restrict__ Y, const bf16* __restrict__ gate,
            const float* __restrict__ normw, bf16* __restrict__ out)
{
  int m = blockIdx.x;
  __shared__ float red[256];
  float vals[INTER / 256];
  float ss = 0.f;
#pragma unroll
  for (int i = 0; i < INTER / 256; ++i) {
    int j = threadIdx.x + i * 256;
    float y = (float)Y[(size_t)m * INTER + j];
    float g = (float)gate[(size_t)m * INTER + j];
    float v = y * silu(g);
    vals[i] = v;
    ss += v * v;
  }
  red[threadIdx.x] = ss;
  __syncthreads();
  for (int off = 128; off > 0; off >>= 1) {
    if (threadIdx.x < off) red[threadIdx.x] += red[threadIdx.x + off];
    __syncthreads();
  }
  float scale = rsqrtf(red[0] / (float)INTER + EPSV);
#pragma unroll
  for (int i = 0; i < INTER / 256; ++i) {
    int j = threadIdx.x + i * 256;
    out[(size_t)m * INTER + j] = (bf16)(vals[i] * scale * normw[j]);
  }
}

// ---------------- host launcher ----------------
extern "C" void kernel_launch(void* const* d_in, const int* in_sizes, int n_in,
                              void* d_out, int out_size, void* d_ws, size_t ws_size,
                              hipStream_t stream) {
  const float* hs      = (const float*)d_in[0];
  const float* w_in    = (const float*)d_in[1];
  const float* conv_w  = (const float*)d_in[2];
  const float* conv_b  = (const float*)d_in[3];
  const float* A_log   = (const float*)d_in[4];
  const float* dt_bias = (const float*)d_in[5];
  const float* Dv      = (const float*)d_in[6];
  const float* norm_w  = (const float*)d_in[7];
  const float* w_out   = (const float*)d_in[8];
  float* out = (float*)d_out;

  char* wsp = (char*)d_ws;
  auto alloc = [&](size_t bytes) -> void* {
    void* p = (void*)wsp;
    wsp += (bytes + 255) & ~(size_t)255;
    return p;
  };
  bf16*  hs_bf   = (bf16*) alloc((size_t)MTOT * HIDDEN * 2);
  bf16*  win_bf  = (bf16*) alloc((size_t)NPROJ_PAD * HIDDEN * 2);
  bf16*  wout_bf = (bf16*) alloc((size_t)HIDDEN * INTER * 2);
  bf16*  gate    = (bf16*) alloc((size_t)MTOT * INTER * 2);
  bf16*  hbc     = (bf16*) alloc((size_t)MTOT * CONV_DIM * 2);
  float* dtraw   = (float*)alloc((size_t)MTOT * HEADS * 4);
  float* dtv     = (float*)alloc((size_t)MTOT * HEADS * 4);
  float* Acum    = (float*)alloc((size_t)BATCH * HEADS * SEQL * 4);
  float* csum    = (float*)alloc((size_t)BATCH * HEADS * NCHUNK * 4);
  bf16*  xs      = (bf16*) alloc((size_t)MTOT * INTER * 2);
  bf16*  xdtT    = (bf16*) alloc((size_t)MTOT * INTER * 2);
  bf16*  Bn      = (bf16*) alloc((size_t)MTOT * STATE * 2);
  bf16*  Cn      = (bf16*) alloc((size_t)MTOT * STATE * 2);
  float* Sloc    = (float*)alloc((size_t)BATCH * NCHUNK * HEADS * HEAD_DIM * STATE * 4);
  bf16*  Sin     = (bf16*) alloc((size_t)BATCH * NCHUNK * HEADS * HEAD_DIM * STATE * 2);
  bf16*  Yb      = (bf16*) alloc((size_t)MTOT * INTER * 2);
  bf16*  ybf     = (bf16*) alloc((size_t)MTOT * INTER * 2);

  // 1) pack bf16 operands
  {
    long n = (long)MTOT * HIDDEN;
    k_f32_to_bf16<<<(n + 255) / 256, 256, 0, stream>>>(hs, hs_bf, n);
  }
  {
    long n = (long)NPROJ_PAD * HIDDEN;
    k_pack_w_in<<<(n + 255) / 256, 256, 0, stream>>>(w_in, win_bf);
  }
  {
    long n = (long)HIDDEN * INTER;
    k_f32_to_bf16<<<(n + 255) / 256, 256, 0, stream>>>(w_out, wout_bf, n);
  }
  // 2) in_proj GEMM (split epilogue)
  k_gemm<<<dim3(NPROJ_PAD / GT_N, MTOT / GT_M), 256, 0, stream>>>(
      hs_bf, win_bf, MTOT, NPROJ_PAD, HIDDEN, 1, nullptr, gate, hbc, dtraw);
  // 3) dt + chunk cumsum
  k_dt<<<BATCH * HEADS * NCHUNK, CHUNK, 0, stream>>>(dtraw, A_log, dt_bias,
                                                     dtv, Acum, csum);
  // 4) conv + silu + operand prep
  k_conv<<<MTOT, 256, 0, stream>>>(hbc, conv_w, conv_b, dtv, xs, xdtT, Bn, Cn);
  // 5) chunk-local states
  k_states<<<BATCH * NCHUNK * HEADS, 256, 0, stream>>>(xdtT, Bn, Acum, Sloc);
  // 6) inter-chunk recurrence
  k_chain<<<BATCH * HEADS * 8, 256, 0, stream>>>(Sloc, csum, Sin);
  // 7) per-chunk Y
  k_ychunk<<<BATCH * NCHUNK * HEADS, 256, 0, stream>>>(Cn, Bn, Sin, xdtT, xs,
                                                       Acum, Dv, Yb);
  // 8) gate + RMS norm
  k_norm<<<MTOT, 256, 0, stream>>>(Yb, gate, norm_w, ybf);
  // 9) out_proj GEMM -> fp32 output
  k_gemm<<<dim3(HIDDEN / GT_N, MTOT / GT_M), 256, 0, stream>>>(
      ybf, wout_bf, MTOT, HIDDEN, INTER, 0, out, nullptr, nullptr, nullptr);
}
